// Speller_65197603553863
// MI455X (gfx1250) — compile-verified, matrix operand order checked by reading, not run on hardware
//
#include <hip/hip_runtime.h>
#include <math.h>

typedef __attribute__((ext_vector_type(16))) _Float16 v16h;
typedef __attribute__((ext_vector_type(8)))  float    v8f;

#define SEQ_S 800
#define TT    250
#define VV    30

// ---------------- WMMA helper (f32 = f16 x f16 + f32, 16x16x32) ----------------

__device__ __forceinline__ v8f wmma32(v16h a, v16h b, v8f c) {
  return __builtin_amdgcn_wmma_f32_16x16x32_f16(false, a, false, b, (short)0, c,
                                                false, false);
}

// A fragment: rowp -> A[row][k0] for this lane's row (row = m0 + (lane&15)).
// elem h: g=h/8, w=h%8 -> K = w + 8*((g<<1)|hi), hi = lane>=16.
__device__ __forceinline__ v16h ld_a_f16(const _Float16* rowp, int lane) {
  int hi = (lane >> 4) & 1;
  v16h a;
#pragma unroll
  for (int g = 0; g < 2; ++g) {
    const _Float16* p = rowp + 8 * ((g << 1) | hi);
#pragma unroll
    for (int w = 0; w < 8; ++w) a[g * 8 + w] = p[w];
  }
  return a;
}

__device__ __forceinline__ v16h ld_a_f32(const float* rowp, int lane) {
  int hi = (lane >> 4) & 1;
  v16h a;
#pragma unroll
  for (int g = 0; g < 2; ++g) {
    const float* p = rowp + 8 * ((g << 1) | hi);
#pragma unroll
    for (int w = 0; w < 8; ++w) a[g * 8 + w] = (_Float16)p[w];
  }
  return a;
}

// Packed-B fragment: P laid out as [ktTile][ntTile][lane][16 halfs] (32B/lane).
__device__ __forceinline__ v16h ld_b_pk(const _Float16* P, int kt, int nt,
                                        int NtTiles, int lane) {
  const _Float16* p = P + ((((size_t)kt * NtTiles + nt) * 32 + lane) << 4);
  return *(const v16h*)p;
}

// ---------------- weight pack: W(KxN f32) -> WMMA B-fragment order (f16) ----------------

__global__ void pack_b(const float* __restrict__ W, _Float16* __restrict__ P,
                       int K, int N) {
  int idx = blockIdx.x * blockDim.x + threadIdx.x;
  if (idx >= K * N) return;
  int h = idx & 15;
  int l = (idx >> 4) & 31;
  int tile = idx >> 9;
  int Nt = N >> 4;
  int kt = tile / Nt, nt = tile - kt * Nt;
  int hi = l >> 4, g = h >> 3, w = h & 7;
  int k = kt * 32 + w + 8 * ((g << 1) | hi);
  int n = nt * 16 + (l & 15);
  P[idx] = (_Float16)W[(size_t)k * N + n];
}

__global__ void cvt_f32_f16(const float* __restrict__ in, _Float16* __restrict__ out,
                            int n) {
  int i = blockIdx.x * blockDim.x + threadIdx.x;
  for (; i < n; i += gridDim.x * blockDim.x) out[i] = (_Float16)in[i];
}

// ---------------- generic GEMM: C = A(f32) @ Bpacked(f16) + bias; 1 wave = 16x64 ----------------

__global__ void gemm_a32_bpk(const float* __restrict__ A, int lda,
                             const _Float16* __restrict__ Bp, int NtTiles,
                             const float* __restrict__ bias,
                             float* __restrict__ C, int ldc, int Mt, int Kt) {
  int wave = (int)((blockIdx.x * blockDim.x + threadIdx.x) >> 5);
  int lane = threadIdx.x & 31;
  int wavesPerM = NtTiles >> 2;
  if (wave >= Mt * wavesPerM) return;  // wave-uniform
  int mt = wave / wavesPerM, ng = wave - mt * wavesPerM;
  int m0 = mt * 16, n0t = ng * 4;
  const float* arow = A + (size_t)(m0 + (lane & 15)) * lda;
  v8f acc0 = {}, acc1 = {}, acc2 = {}, acc3 = {};
  for (int kt = 0; kt < Kt; ++kt) {
    v16h a = ld_a_f32(arow + kt * 32, lane);
    acc0 = wmma32(a, ld_b_pk(Bp, kt, n0t + 0, NtTiles, lane), acc0);
    acc1 = wmma32(a, ld_b_pk(Bp, kt, n0t + 1, NtTiles, lane), acc1);
    acc2 = wmma32(a, ld_b_pk(Bp, kt, n0t + 2, NtTiles, lane), acc2);
    acc3 = wmma32(a, ld_b_pk(Bp, kt, n0t + 3, NtTiles, lane), acc3);
  }
  int col = lane & 15, hi = (lane >> 4) & 1;
  v8f accs[4] = {acc0, acc1, acc2, acc3};
#pragma unroll
  for (int i = 0; i < 4; ++i) {
    int n = (n0t + i) * 16 + col;
#pragma unroll
    for (int r = 0; r < 8; ++r)
      C[(size_t)(m0 + r + 8 * hi) * ldc + n] = accs[i][r] + bias[n];
  }
}

// ---------------- LSTM1 gates: [emb(y) | ctx | h1] @ [Wih1 ; Whh1] + biases ----------------
// M=64, N=2048, K=512+256+512; 1 wave = 16x64 -> 128 waves

__global__ void gates1_kernel(const _Float16* __restrict__ emb16,   // 30 x 512 (row f16)
                              const float* __restrict__ ctx,        // 64 x 256
                              const float* __restrict__ h1,         // 64 x 512
                              const _Float16* __restrict__ Wih1p,   // packed 24x128 tiles
                              const _Float16* __restrict__ Whh1p,   // packed 16x128 tiles
                              const float* __restrict__ b_ih1,
                              const float* __restrict__ b_hh1,
                              const int* __restrict__ y, int t,
                              float* __restrict__ gates) {
  int wave = (int)((blockIdx.x * blockDim.x + threadIdx.x) >> 5);
  int lane = threadIdx.x & 31;
  int mt = wave >> 5, ng = wave & 31;  // 32 n-groups of 4 tiles
  int m0 = mt * 16, n0t = ng * 4;
  int brow = m0 + (lane & 15);
  int tok = (t == 0) ? 0 : y[brow * TT + (t - 1)];
  const _Float16* erow = emb16 + (size_t)tok * 512;
  const float* crow = ctx + (size_t)brow * 256;
  const float* hrow = h1 + (size_t)brow * 512;
  v8f acc0 = {}, acc1 = {}, acc2 = {}, acc3 = {};
#pragma unroll 2
  for (int kt = 0; kt < 16; ++kt) {  // embedding segment (K 0..511)
    v16h a = ld_a_f16(erow + kt * 32, lane);
    acc0 = wmma32(a, ld_b_pk(Wih1p, kt, n0t + 0, 128, lane), acc0);
    acc1 = wmma32(a, ld_b_pk(Wih1p, kt, n0t + 1, 128, lane), acc1);
    acc2 = wmma32(a, ld_b_pk(Wih1p, kt, n0t + 2, 128, lane), acc2);
    acc3 = wmma32(a, ld_b_pk(Wih1p, kt, n0t + 3, 128, lane), acc3);
  }
#pragma unroll 2
  for (int kt = 0; kt < 8; ++kt) {   // context segment (K 512..767)
    v16h a = ld_a_f32(crow + kt * 32, lane);
    acc0 = wmma32(a, ld_b_pk(Wih1p, 16 + kt, n0t + 0, 128, lane), acc0);
    acc1 = wmma32(a, ld_b_pk(Wih1p, 16 + kt, n0t + 1, 128, lane), acc1);
    acc2 = wmma32(a, ld_b_pk(Wih1p, 16 + kt, n0t + 2, 128, lane), acc2);
    acc3 = wmma32(a, ld_b_pk(Wih1p, 16 + kt, n0t + 3, 128, lane), acc3);
  }
#pragma unroll 2
  for (int kt = 0; kt < 16; ++kt) {  // hidden segment
    v16h a = ld_a_f32(hrow + kt * 32, lane);
    acc0 = wmma32(a, ld_b_pk(Whh1p, kt, n0t + 0, 128, lane), acc0);
    acc1 = wmma32(a, ld_b_pk(Whh1p, kt, n0t + 1, 128, lane), acc1);
    acc2 = wmma32(a, ld_b_pk(Whh1p, kt, n0t + 2, 128, lane), acc2);
    acc3 = wmma32(a, ld_b_pk(Whh1p, kt, n0t + 3, 128, lane), acc3);
  }
  int col = lane & 15, hi = (lane >> 4) & 1;
  v8f accs[4] = {acc0, acc1, acc2, acc3};
#pragma unroll
  for (int i = 0; i < 4; ++i) {
    int n = (n0t + i) * 16 + col;
#pragma unroll
    for (int r = 0; r < 8; ++r)
      gates[(size_t)(m0 + r + 8 * hi) * 2048 + n] = accs[i][r] + b_ih1[n] + b_hh1[n];
  }
}

// ---------------- LSTM2 gates: [h1 | h2] @ [Wih2 ; Whh2] + biases ----------------
// M=64, N=1024, K=512+256; 64 waves

__global__ void gates2_kernel(const float* __restrict__ h1,
                              const float* __restrict__ h2,
                              const _Float16* __restrict__ Wih2p,  // 16x64 tiles
                              const _Float16* __restrict__ Whh2p,  // 8x64 tiles
                              const float* __restrict__ b_ih2,
                              const float* __restrict__ b_hh2,
                              float* __restrict__ gates) {
  int wave = (int)((blockIdx.x * blockDim.x + threadIdx.x) >> 5);
  int lane = threadIdx.x & 31;
  int mt = wave >> 4, ng = wave & 15;  // 16 n-groups of 4 tiles
  int m0 = mt * 16, n0t = ng * 4;
  int brow = m0 + (lane & 15);
  const float* h1row = h1 + (size_t)brow * 512;
  const float* h2row = h2 + (size_t)brow * 256;
  v8f acc0 = {}, acc1 = {}, acc2 = {}, acc3 = {};
#pragma unroll 2
  for (int kt = 0; kt < 16; ++kt) {
    v16h a = ld_a_f32(h1row + kt * 32, lane);
    acc0 = wmma32(a, ld_b_pk(Wih2p, kt, n0t + 0, 64, lane), acc0);
    acc1 = wmma32(a, ld_b_pk(Wih2p, kt, n0t + 1, 64, lane), acc1);
    acc2 = wmma32(a, ld_b_pk(Wih2p, kt, n0t + 2, 64, lane), acc2);
    acc3 = wmma32(a, ld_b_pk(Wih2p, kt, n0t + 3, 64, lane), acc3);
  }
#pragma unroll 2
  for (int kt = 0; kt < 8; ++kt) {
    v16h a = ld_a_f32(h2row + kt * 32, lane);
    acc0 = wmma32(a, ld_b_pk(Whh2p, kt, n0t + 0, 64, lane), acc0);
    acc1 = wmma32(a, ld_b_pk(Whh2p, kt, n0t + 1, 64, lane), acc1);
    acc2 = wmma32(a, ld_b_pk(Whh2p, kt, n0t + 2, 64, lane), acc2);
    acc3 = wmma32(a, ld_b_pk(Whh2p, kt, n0t + 3, 64, lane), acc3);
  }
  int col = lane & 15, hi = (lane >> 4) & 1;
  v8f accs[4] = {acc0, acc1, acc2, acc3};
#pragma unroll
  for (int i = 0; i < 4; ++i) {
    int n = (n0t + i) * 16 + col;
#pragma unroll
    for (int r = 0; r < 8; ++r)
      gates[(size_t)(m0 + r + 8 * hi) * 1024 + n] = accs[i][r] + b_ih2[n] + b_hh2[n];
  }
}

// ---------------- LSTM cell elementwise ----------------

__device__ __forceinline__ float sigmoidf_(float x) { return 1.0f / (1.0f + __expf(-x)); }

__global__ void lstm_cell(const float* __restrict__ gates, float* __restrict__ h,
                          float* __restrict__ c, int U) {
  int idx = blockIdx.x * blockDim.x + threadIdx.x;
  if (idx >= 64 * U) return;
  int b = idx / U, u = idx - b * U;
  const float* g = gates + (size_t)b * 4 * U;
  float ig = sigmoidf_(g[u]);
  float fg = sigmoidf_(g[U + u]);
  float gg = tanhf(g[2 * U + u]);
  float og = sigmoidf_(g[3 * U + u]);
  float cn = fg * c[idx] + ig * gg;
  c[idx] = cn;
  h[idx] = og * tanhf(cn);
}

// ---------------- fused attention: q-proj -> scores -> softmax -> context -> logits ----------------
// one workgroup (256 threads) per batch row

__global__ void attention_kernel(const float* __restrict__ key_proj,
                                 const float* __restrict__ val_proj,
                                 const float* __restrict__ h2,
                                 const float* __restrict__ Wq,   // 256 x 256 f32
                                 const float* __restrict__ bq,
                                 const int* __restrict__ lens,
                                 const float* __restrict__ embed_W,  // 30 x 512 f32
                                 const float* __restrict__ char_b,
                                 float* __restrict__ ctx,
                                 float* __restrict__ pred,
                                 float* __restrict__ attn_plot, int t) {
  __shared__ float hs[256];
  __shared__ float qs[256];
  __shared__ float cxs[256];
  __shared__ float sc[SEQ_S];
  __shared__ float red[256];
  int b = blockIdx.x, tid = threadIdx.x;
  hs[tid] = h2[b * 256 + tid];
  __syncthreads();
  // q = h2 @ Wq + bq  (Wq reads coalesced across tid)
  float qa = bq[tid];
#pragma unroll 8
  for (int j = 0; j < 256; ++j) qa += hs[j] * Wq[(size_t)j * 256 + tid];
  qs[tid] = qa;
  __syncthreads();
  int len = lens[b];
  for (int s = tid; s < SEQ_S; s += 256) {
    const float* kp = key_proj + ((size_t)b * SEQ_S + s) * 256;
    float acc = 0.0f;
#pragma unroll 8
    for (int p = 0; p < 256; ++p) acc += kp[p] * qs[p];
    sc[s] = (s < len) ? acc * 0.0625f : -1.0e9f;
  }
  __syncthreads();
  float m = -3.4e38f;
  for (int s = tid; s < SEQ_S; s += 256) m = fmaxf(m, sc[s]);
  red[tid] = m;
  __syncthreads();
  for (int off = 128; off > 0; off >>= 1) {
    if (tid < off) red[tid] = fmaxf(red[tid], red[tid + off]);
    __syncthreads();
  }
  m = red[0];
  __syncthreads();
  float sum = 0.0f;
  for (int s = tid; s < SEQ_S; s += 256) {
    float e = __expf(sc[s] - m);
    sc[s] = e;
    sum += e;
  }
  red[tid] = sum;
  __syncthreads();
  for (int off = 128; off > 0; off >>= 1) {
    if (tid < off) red[tid] += red[tid + off];
    __syncthreads();
  }
  float inv = 1.0f / red[0];
  for (int s = tid; s < SEQ_S; s += 256) sc[s] *= inv;
  __syncthreads();
  if (b == 0)
    for (int s = tid; s < SEQ_S; s += 256) attn_plot[(size_t)s * TT + t] = sc[s];
  // context[p] = sum_s aw[s] * val_proj[b, s, p]; p == tid
  const float* vp = val_proj + (size_t)b * SEQ_S * 256 + tid;
  float acc = 0.0f;
  for (int s = 0; s < SEQ_S; ++s) {
    __builtin_prefetch(vp + (size_t)(s + 8) * 256, 0, 1);
    acc += sc[s] * vp[(size_t)s * 256];
  }
  ctx[b * 256 + tid] = acc;
  cxs[tid] = acc;
  __syncthreads();
  // logits = [h2 | ctx] @ embed_W^T + char_b
  if (tid < VV) {
    const float* er = embed_W + (size_t)tid * 512;
    float lg = char_b[tid];
#pragma unroll 8
    for (int j = 0; j < 256; ++j) lg += hs[j] * er[j];
#pragma unroll 8
    for (int j = 0; j < 256; ++j) lg += cxs[j] * er[256 + j];
    pred[((size_t)b * TT + t) * VV + tid] = lg;
  }
}

// ---------------- init: zero states, ctx = key_proj[:, 0, :] ----------------

__global__ void init_state(float* h1, float* c1, float* h2, float* c2, float* ctx,
                           const float* __restrict__ key_proj) {
  int i = blockIdx.x * blockDim.x + threadIdx.x;
  if (i < 64 * 512) { h1[i] = 0.0f; c1[i] = 0.0f; }
  if (i < 64 * 256) {
    h2[i] = 0.0f; c2[i] = 0.0f;
    int b = i >> 8, p = i & 255;
    ctx[i] = key_proj[(size_t)b * SEQ_S * 256 + p];
  }
}

// ---------------- workspace layout (bytes, all 256B aligned) ----------------
constexpr size_t KP_OFF    = 0;                        // key_proj 64*800*256 f32
constexpr size_t VP_OFF    = KP_OFF    + 52428800;     // val_proj
constexpr size_t WIH1_OFF  = VP_OFF    + 52428800;     // 768*2048 f16 packed
constexpr size_t WHH1_OFF  = WIH1_OFF  + 3145728;      // 512*2048 f16 packed
constexpr size_t WIH2_OFF  = WHH1_OFF  + 2097152;      // 512*1024 f16 packed
constexpr size_t WHH2_OFF  = WIH2_OFF  + 1048576;      // 256*1024 f16 packed
constexpr size_t WK_OFF    = WHH2_OFF  + 524288;       // 512*256 f16 packed
constexpr size_t WV_OFF    = WK_OFF    + 262144;       // 512*256 f16 packed
constexpr size_t EMB_OFF   = WV_OFF    + 262144;       // 30*512 f16 row-major
constexpr size_t G1_OFF    = EMB_OFF   + 30720;        // 64*2048 f32
constexpr size_t G2_OFF    = G1_OFF    + 524288;       // 64*1024 f32
constexpr size_t H1_OFF    = G2_OFF    + 262144;       // 64*512 f32
constexpr size_t C1_OFF    = H1_OFF    + 131072;
constexpr size_t H2_OFF    = C1_OFF    + 131072;       // 64*256 f32
constexpr size_t C2_OFF    = H2_OFF    + 65536;
constexpr size_t CTX_OFF   = C2_OFF    + 65536;

extern "C" void kernel_launch(void* const* d_in, const int* in_sizes, int n_in,
                              void* d_out, int out_size, void* d_ws, size_t ws_size,
                              hipStream_t stream) {
  const float* enc    = (const float*)d_in[0];
  const float* embedW = (const float*)d_in[1];
  const float* W_ih1  = (const float*)d_in[2];
  const float* b_ih1  = (const float*)d_in[3];
  const float* W_hh1  = (const float*)d_in[4];
  const float* b_hh1  = (const float*)d_in[5];
  const float* W_ih2  = (const float*)d_in[6];
  const float* b_ih2  = (const float*)d_in[7];
  const float* W_hh2  = (const float*)d_in[8];
  const float* b_hh2  = (const float*)d_in[9];
  const float* Wk     = (const float*)d_in[10];
  const float* bk     = (const float*)d_in[11];
  const float* Wv     = (const float*)d_in[12];
  const float* bv     = (const float*)d_in[13];
  const float* Wq     = (const float*)d_in[14];
  const float* bq     = (const float*)d_in[15];
  const float* char_b = (const float*)d_in[16];
  const int*   lens   = (const int*)d_in[17];
  const int*   y      = (const int*)d_in[18];

  char* ws = (char*)d_ws;
  float*    key_proj = (float*)(ws + KP_OFF);
  float*    val_proj = (float*)(ws + VP_OFF);
  _Float16* wih1p    = (_Float16*)(ws + WIH1_OFF);
  _Float16* whh1p    = (_Float16*)(ws + WHH1_OFF);
  _Float16* wih2p    = (_Float16*)(ws + WIH2_OFF);
  _Float16* whh2p    = (_Float16*)(ws + WHH2_OFF);
  _Float16* wkp      = (_Float16*)(ws + WK_OFF);
  _Float16* wvp      = (_Float16*)(ws + WV_OFF);
  _Float16* emb16    = (_Float16*)(ws + EMB_OFF);
  float*    gates1   = (float*)(ws + G1_OFF);
  float*    gates2   = (float*)(ws + G2_OFF);
  float*    h1       = (float*)(ws + H1_OFF);
  float*    c1       = (float*)(ws + C1_OFF);
  float*    h2       = (float*)(ws + H2_OFF);
  float*    c2       = (float*)(ws + C2_OFF);
  float*    ctx      = (float*)(ws + CTX_OFF);

  float* pred = (float*)d_out;                // 64 x 250 x 30
  float* plot = pred + (size_t)64 * TT * VV;  // 800 x 250

  auto pack = [&](const float* src, _Float16* dst, int K, int N) {
    int n = K * N;
    pack_b<<<(n + 255) / 256, 256, 0, stream>>>(src, dst, K, N);
  };
  pack(W_ih1, wih1p, 768, 2048);
  pack(W_hh1, whh1p, 512, 2048);
  pack(W_ih2, wih2p, 512, 1024);
  pack(W_hh2, whh2p, 256, 1024);
  pack(Wk, wkp, 512, 256);
  pack(Wv, wvp, 512, 256);
  cvt_f32_f16<<<60, 256, 0, stream>>>(embedW, emb16, 30 * 512);

  // key/val projections: Mt=3200, NtTiles=16 -> 12800 waves -> 1600 blocks
  gemm_a32_bpk<<<1600, 256, 0, stream>>>(enc, 512, wkp, 16, bk, key_proj, 256,
                                         3200, 16);
  gemm_a32_bpk<<<1600, 256, 0, stream>>>(enc, 512, wvp, 16, bv, val_proj, 256,
                                         3200, 16);

  init_state<<<128, 256, 0, stream>>>(h1, c1, h2, c2, ctx, key_proj);

  for (int t = 0; t < TT; ++t) {
    gates1_kernel<<<16, 256, 0, stream>>>(emb16, ctx, h1, wih1p, whh1p,
                                          b_ih1, b_hh1, y, t, gates1);  // 128 waves
    lstm_cell<<<128, 256, 0, stream>>>(gates1, h1, c1, 512);
    gates2_kernel<<<8, 256, 0, stream>>>(h1, h2, wih2p, whh2p,
                                         b_ih2, b_hh2, gates2);         // 64 waves
    lstm_cell<<<64, 256, 0, stream>>>(gates2, h2, c2, 256);
    attention_kernel<<<64, 256, 0, stream>>>(key_proj, val_proj, h2, Wq, bq, lens,
                                             embedW, char_b, ctx, pred, plot, t);
  }
}
